// LSTM_19267223290530
// MI455X (gfx1250) — matrix operand order repaired and checked by reading, not compiled, not run on hardware
//
#include <hip/hip_runtime.h>
#include <math.h>

// ---------------------------------------------------------------------------
// 2-layer LSTM (B=64,T=1024,IN=128,H=512) + FC(512->64) for gfx1250 (CDNA5).
// Persistent 16-WG kernel, f16 WMMA (v_wmma_f32_16x16x32_f16), async
// global->LDS staging of the hidden state (ASYNCcnt), device-scope barrier
// per half-step, cell state in registers.
// ---------------------------------------------------------------------------

typedef _Float16 v16h  __attribute__((ext_vector_type(16)));
typedef _Float16 half8 __attribute__((ext_vector_type(8)));
typedef float    v8f   __attribute__((ext_vector_type(8)));

#define B_   64
#define T_   1024
#define IN_  128
#define H_   512
#define G_   2048     // 4*H
#define OUT_ 64
#define NWG  16
#define NTH  512      // 16 wave32 per workgroup

// ---------------- WMMA fragment loaders (per CDNA5 ISA VGPR layouts) -------
// A (16x32 f16): lane holds row m=lane&15; halves k = hi*8+0..7 and 16+hi*8+0..7
__device__ __forceinline__ v16h load_a_frag(const _Float16* p, int hi) {
    const half8* q = (const half8*)(p + (hi << 3));
    half8 lo = q[0];
    half8 h2 = q[2];            // +16 halves
    v16h v;
#pragma unroll
    for (int i = 0; i < 8; ++i) { v[i] = lo[i]; v[i + 8] = h2[i]; }
    return v;
}
// B (32x16 f16): lane holds column n=lane&15; halves k = hi*8+0..7 and 16+hi*8+0..7
__device__ __forceinline__ v16h load_b_frag(const _Float16* p, int hi) {
    const half8* q = (const half8*)(p + (hi << 3));
    half8 lo = q[0];
    half8 h2 = q[2];
    v16h v;
#pragma unroll
    for (int i = 0; i < 8; ++i) { v[i] = lo[i]; v[i + 8] = h2[i]; }
    return v;
}

// Two output tiles (same N-tile, two M-tiles) sharing the B fragment.
// D[m][n] += sum_k A[m][k] * W[n][k]   (i.e. A @ W^T)
__device__ __forceinline__ void mma_block2(
    v8f& acc0, v8f& acc1,
    const _Float16* __restrict A, int lda, int m0, int m1,
    const _Float16* __restrict W, int ldw, int n0,
    int kTiles, int ln, int hi)
{
    const _Float16* pa0 = A + (m0 + ln) * lda;
    const _Float16* pa1 = A + (m1 + ln) * lda;
    const _Float16* pb  = W + (n0 + ln) * ldw;
#pragma unroll 2
    for (int kt = 0; kt < kTiles; ++kt) {
        v16h bf = load_b_frag(pb,  hi);
        v16h a0 = load_a_frag(pa0, hi);
        v16h a1 = load_a_frag(pa1, hi);
        acc0 = __builtin_amdgcn_wmma_f32_16x16x32_f16(false, a0, false, bf,
                                                      (short)0, acc0, false, false);
        acc1 = __builtin_amdgcn_wmma_f32_16x16x32_f16(false, a1, false, bf,
                                                      (short)0, acc1, false, false);
        pa0 += 32; pa1 += 32; pb += 32;
    }
}

__device__ __forceinline__ void mma_tile(
    v8f& acc,
    const _Float16* __restrict A, int lda, int m0,
    const _Float16* __restrict W, int ldw, int n0,
    int kTiles, int ln, int hi)
{
    const _Float16* pa = A + (m0 + ln) * lda;
    const _Float16* pb = W + (n0 + ln) * ldw;
#pragma unroll 2
    for (int kt = 0; kt < kTiles; ++kt) {
        v16h a = load_a_frag(pa, hi);
        v16h b = load_b_frag(pb, hi);
        acc = __builtin_amdgcn_wmma_f32_16x16x32_f16(false, a, false, b,
                                                     (short)0, acc, false, false);
        pa += 32; pb += 32;
    }
}

// ---------------- async global -> LDS staging (CDNA5 ASYNCcnt path) --------
// Copies 64KB (B_*H_ halves) from global `src` into LDS offset 0.
// Each thread moves 8x16B chunks; data never touches VGPRs.
__device__ __forceinline__ void stage_async(const _Float16* src, int tid) {
#pragma unroll
    for (int i = 0; i < 8; ++i) {
        unsigned lds_off = (unsigned)((tid + i * NTH) * 16);
        unsigned long long g =
            (unsigned long long)(const char*)src + (unsigned long long)(tid + i * NTH) * 16ull;
        asm volatile("global_load_async_to_lds_b128 %0, %1, off"
                     :: "v"(lds_off), "v"(g)
                     : "memory");
    }
    asm volatile("s_wait_asynccnt 0x0" ::: "memory");
}

// ---------------- device-scope (grid) barrier, sense-reversing -------------
__device__ __forceinline__ void grid_barrier(int* cnt, int* gen) {
    __syncthreads();
    if (threadIdx.x == 0) {
        __threadfence();
        volatile int* vgen = gen;
        int g = *vgen;
        int a = atomicAdd(cnt, 1);
        if (a == NWG - 1) {
            atomicExch(cnt, 0);
            __threadfence();
            atomicAdd(gen, 1);
        } else {
            while (*vgen == g) { __builtin_amdgcn_s_sleep(2); }
        }
        __threadfence();
    }
    __syncthreads();
}

__device__ __forceinline__ float sigmoidf_(float x) {
    return 1.0f / (1.0f + __expf(-x));
}

// ---------------- prep kernels ---------------------------------------------
__global__ void k_cvt_f16(const float* __restrict s, _Float16* __restrict d, int n) {
    int i = blockIdx.x * blockDim.x + threadIdx.x;
    if (i < n) d[i] = (_Float16)s[i];
}
__global__ void k_bias_sum(const float* __restrict a, const float* __restrict b,
                           float* __restrict d, int n) {
    int i = blockIdx.x * blockDim.x + threadIdx.x;
    if (i < n) d[i] = a[i] + b[i];
}

// ---------------- persistent LSTM kernel -----------------------------------
__global__ void __launch_bounds__(NTH, 1)
k_lstm_persistent(const _Float16* __restrict x16,
                  const _Float16* __restrict wih0, const _Float16* __restrict whh0,
                  const _Float16* __restrict wih1, const _Float16* __restrict whh1,
                  const _Float16* __restrict fcw,
                  const float* __restrict bias0, const float* __restrict bias1,
                  const float* __restrict fcb,
                  _Float16* __restrict h1buf,   // [2][B_][H_] ping-pong
                  _Float16* __restrict h2buf,   // [2][B_][H_] ping-pong
                  int* bar_cnt, int* bar_gen,
                  float* __restrict out)
{
    __shared__ __align__(16) _Float16 smem[B_ * H_];   // 64 KB h-stage (LDS off 0)
    float* gbuf = (float*)smem;                        // aliased 32 KB gate buf

    const int tid  = threadIdx.x;
    const int wg   = blockIdx.x;
    const int lane = tid & 31;
    const int wid  = tid >> 5;
    const int ln   = lane & 15;
    const int hi   = lane >> 4;

    // 32 tiles per WG = [4 mtiles][8 ntiles]; wave wid owns tiles wid and wid+16
    const int nt    = wid & 7;
    const int mt0   = (wid >> 3) * 16;       // row 0 or 16
    const int mt1   = mt0 + 32;              // row 32 or 48
    const int gate  = nt >> 1;
    const int nglob = gate * H_ + wg * 32 + (nt & 1) * 16;  // weight-row base

    // zero ping buffers (index 0) of h1/h2: 8192 16B chunks across the grid
    {
        int g = wg * NTH + tid;              // 0..8191
        half8 z;
#pragma unroll
        for (int i = 0; i < 8; ++i) z[i] = (_Float16)0.0f;
        if (g < 4096) ((half8*)h1buf)[g] = z;
        else          ((half8*)h2buf)[g - 4096] = z;
    }

    float c0reg[4], c1reg[4];                // cell state lives in registers
#pragma unroll
    for (int s = 0; s < 4; ++s) { c0reg[s] = 0.0f; c1reg[s] = 0.0f; }

    grid_barrier(bar_cnt, bar_gen);

    for (int t = 0; t < T_; ++t) {
        const int p = t & 1;
        _Float16* h1prev = h1buf + (size_t)p * (B_ * H_);
        _Float16* h1cur  = h1buf + (size_t)(p ^ 1) * (B_ * H_);
        _Float16* h2prev = h2buf + (size_t)p * (B_ * H_);
        _Float16* h2cur  = h2buf + (size_t)(p ^ 1) * (B_ * H_);

        // ================= layer 0 =================
        stage_async(h1prev, tid);             // h1_{t-1} -> LDS (async, no VGPRs)
        __syncthreads();

        v8f a0 = {0.f,0.f,0.f,0.f,0.f,0.f,0.f,0.f};
        v8f a1 = a0;
        // recurrent term: h1_{t-1} (LDS) @ w_hh0^T
        mma_block2(a0, a1, smem, H_, mt0, mt1, whh0, H_, nglob, H_ / 32, ln, hi);
        // input term: x_t (global, L2-resident) @ w_ih0^T
        mma_block2(a0, a1, x16 + (size_t)t * IN_, T_ * IN_, mt0, mt1,
                   wih0, IN_, nglob, IN_ / 32, ln, hi);

        __syncthreads();                      // before aliasing LDS as gate buf
#pragma unroll
        for (int j = 0; j < 8; ++j) {
            gbuf[(mt0 + hi * 8 + j) * 128 + nt * 16 + ln] = a0[j];
            gbuf[(mt1 + hi * 8 + j) * 128 + nt * 16 + ln] = a1[j];
        }
        __syncthreads();

#pragma unroll
        for (int s = 0; s < 4; ++s) {
            int e   = tid + s * NTH;          // 0..2047 over [64 b][32 hc]
            int b   = e >> 5;
            int hc  = e & 31;
            int col = wg * 32 + hc;
            float gi = sigmoidf_(gbuf[b * 128 +       hc] + bias0[          col]);
            float gf = sigmoidf_(gbuf[b * 128 + 32 +  hc] + bias0[H_     + col]);
            float gg = tanhf    (gbuf[b * 128 + 64 +  hc] + bias0[2 * H_ + col]);
            float go = sigmoidf_(gbuf[b * 128 + 96 +  hc] + bias0[3 * H_ + col]);
            float c  = gf * c0reg[s] + gi * gg;
            c0reg[s] = c;
            h1cur[b * H_ + col] = (_Float16)(go * tanhf(c));
        }
        grid_barrier(bar_cnt, bar_gen);

        // ================= layer 1 =================
        stage_async(h2prev, tid);             // recurrent input first
        __syncthreads();
        v8f b0 = {0.f,0.f,0.f,0.f,0.f,0.f,0.f,0.f};
        v8f b1 = b0;
        mma_block2(b0, b1, smem, H_, mt0, mt1, whh1, H_, nglob, H_ / 32, ln, hi);
        __syncthreads();
        stage_async(h1cur, tid);              // layer-1 input term: h1_t
        __syncthreads();
        mma_block2(b0, b1, smem, H_, mt0, mt1, wih1, H_, nglob, H_ / 32, ln, hi);

        __syncthreads();
#pragma unroll
        for (int j = 0; j < 8; ++j) {
            gbuf[(mt0 + hi * 8 + j) * 128 + nt * 16 + ln] = b0[j];
            gbuf[(mt1 + hi * 8 + j) * 128 + nt * 16 + ln] = b1[j];
        }
        __syncthreads();

#pragma unroll
        for (int s = 0; s < 4; ++s) {
            int e   = tid + s * NTH;
            int b   = e >> 5;
            int hc  = e & 31;
            int col = wg * 32 + hc;
            float gi = sigmoidf_(gbuf[b * 128 +       hc] + bias1[          col]);
            float gf = sigmoidf_(gbuf[b * 128 + 32 +  hc] + bias1[H_     + col]);
            float gg = tanhf    (gbuf[b * 128 + 64 +  hc] + bias1[2 * H_ + col]);
            float go = sigmoidf_(gbuf[b * 128 + 96 +  hc] + bias1[3 * H_ + col]);
            float c  = gf * c1reg[s] + gi * gg;
            c1reg[s] = c;
            h2cur[b * H_ + col] = (_Float16)(go * tanhf(c));
        }
        grid_barrier(bar_cnt, bar_gen);
    }

    // ================= final FC: out = h2_last @ fc_w^T + fc_b =============
    // T_ even -> last h2cur index = buffer 0
    if (wg == 0) {
        const _Float16* hlast = h2buf;        // buffer 0
        v8f acc = {0.f,0.f,0.f,0.f,0.f,0.f,0.f,0.f};
        int mt  = (wid >> 2) * 16;            // 0,16,32,48
        int ntt = (wid & 3) * 16;             // 0,16,32,48
        mma_tile(acc, hlast, H_, mt, fcw, H_, ntt, H_ / 32, ln, hi);
#pragma unroll
        for (int j = 0; j < 8; ++j) {
            int b = mt + hi * 8 + j;
            int n = ntt + ln;
            out[b * OUT_ + n] = acc[j] + fcb[n];
        }
    }
}

// ---------------------------------------------------------------------------
extern "C" void kernel_launch(void* const* d_in, const int* in_sizes, int n_in,
                              void* d_out, int out_size, void* d_ws, size_t ws_size,
                              hipStream_t stream)
{
    (void)in_sizes; (void)n_in; (void)out_size; (void)ws_size;

    const float* x     = (const float*)d_in[0];
    const float* wih0f = (const float*)d_in[1];
    const float* whh0f = (const float*)d_in[2];
    const float* bih0  = (const float*)d_in[3];
    const float* bhh0  = (const float*)d_in[4];
    const float* wih1f = (const float*)d_in[5];
    const float* whh1f = (const float*)d_in[6];
    const float* bih1  = (const float*)d_in[7];
    const float* bhh1  = (const float*)d_in[8];
    const float* fcwf  = (const float*)d_in[9];
    const float* fcb   = (const float*)d_in[10];
    float* out = (float*)d_out;

    char*  ws  = (char*)d_ws;
    size_t off = 0;
    auto carve = [&](size_t bytes) -> void* {
        void* p = ws + off;
        off = (off + bytes + 255) & ~(size_t)255;
        return p;
    };

    _Float16* x16   = (_Float16*)carve((size_t)B_ * T_ * IN_ * 2);   // 16 MB
    _Float16* wih0  = (_Float16*)carve((size_t)G_ * IN_ * 2);
    _Float16* whh0  = (_Float16*)carve((size_t)G_ * H_ * 2);
    _Float16* wih1  = (_Float16*)carve((size_t)G_ * H_ * 2);
    _Float16* whh1  = (_Float16*)carve((size_t)G_ * H_ * 2);
    _Float16* fcw   = (_Float16*)carve((size_t)OUT_ * H_ * 2);
    float*    bias0 = (float*)carve((size_t)G_ * 4);
    float*    bias1 = (float*)carve((size_t)G_ * 4);
    _Float16* h1buf = (_Float16*)carve((size_t)2 * B_ * H_ * 2);
    _Float16* h2buf = (_Float16*)carve((size_t)2 * B_ * H_ * 2);
    int*      bar   = (int*)carve(256);

    auto cvt = [&](const float* s, _Float16* d, int n) {
        k_cvt_f16<<<(n + 255) / 256, 256, 0, stream>>>(s, d, n);
    };
    cvt(x,     x16,  B_ * T_ * IN_);
    cvt(wih0f, wih0, G_ * IN_);
    cvt(whh0f, whh0, G_ * H_);
    cvt(wih1f, wih1, G_ * H_);
    cvt(whh1f, whh1, G_ * H_);
    cvt(fcwf,  fcw,  OUT_ * H_);
    k_bias_sum<<<(G_ + 255) / 256, 256, 0, stream>>>(bih0, bhh0, bias0, G_);
    k_bias_sum<<<(G_ + 255) / 256, 256, 0, stream>>>(bih1, bhh1, bias1, G_);

    hipMemsetAsync(bar, 0, 2 * sizeof(int), stream);

    k_lstm_persistent<<<NWG, NTH, 0, stream>>>(
        x16, wih0, whh0, wih1, whh1, fcw, bias0, bias1, fcb,
        h1buf, h2buf, bar + 0, bar + 1, out);
}
